// PerceiverResampler_25434796327655
// MI455X (gfx1250) — matrix operand order repaired and checked by reading, not compiled, not run on hardware
//
#include <hip/hip_runtime.h>

// ---------------------------------------------------------------------------
// PerceiverResampler for MI455X (gfx1250, wave32, WMMA bf16 16x16x32)
//
//  * x normalized ONCE -> xhat bf16; per-layer LN affine folded into Wkv.
//  * GEMMs: v_wmma_f32_16x16x32_bf16, double-buffered LDS staging:
//      - A panel: global_load_async_to_lds_b128   (ASYNCcnt)
//      - B panel: TDM tensor_load_to_lds 2D tile  (TENSORcnt, wave 0 only,
//        guarded by a true scalar branch since TDM ignores EXEC)
//  * Attention: online-softmax VALU kernel (5% of FLOPs, streaming K/V).
// ---------------------------------------------------------------------------

typedef __attribute__((ext_vector_type(16))) __bf16 v16bf;
typedef __attribute__((ext_vector_type(8)))  __bf16 v8bf;
typedef __attribute__((ext_vector_type(8)))  float  v8f;
typedef __attribute__((ext_vector_type(4)))  unsigned u32x4;
typedef __attribute__((ext_vector_type(8)))  unsigned u32x8;

#define DIM   1024
#define DEPTH 6
#define DH    64
#define NHEAD 8
#define NLAT  64
#define INNER 512
#define FFDIM 4096
#define NMEDIA 4096           // F*v = 16*256 tokens per (b,T)
#define NBT    16             // b*T
#define XROWS  (NBT * NMEDIA) // 65536
#define LROWS  (NBT * NLAT)   // 1024

// GEMM tiling
#define BM 128
#define BN 64
#define BK 64
#define A_ELEMS (BM * BK)     // 8192 bf16 = 16KB
#define B_ELEMS (BN * BK)     // 4096 bf16 = 8KB

// ---------------------------------------------------------------------------
// CDNA5 async-copy / TDM primitives (inline asm: portable across toolchains)
// ---------------------------------------------------------------------------
__device__ __forceinline__ void async_copy_b128(const __bf16* gsrc, __bf16* lds_dst) {
  unsigned lds_off = (unsigned)(uintptr_t)lds_dst;              // low 32 bits = LDS byte addr
  unsigned long long gaddr = (unsigned long long)(uintptr_t)gsrc;
  asm volatile("global_load_async_to_lds_b128 %0, %1, off"
               :: "v"(lds_off), "v"(gaddr) : "memory");
}
__device__ __forceinline__ void wait_async_le4() {
  asm volatile("s_wait_asynccnt 0x4" ::: "memory");
}
__device__ __forceinline__ void wait_async_0() {
  asm volatile("s_wait_asynccnt 0x0" ::: "memory");
}

// TDM: load a 2D tile (BK elems x BN rows, bf16, row stride ldbt elems)
// from global into LDS. D# per CDNA5 ISA (groups 0/1; 2/3 NULL).
__device__ __forceinline__ void tdm_load_b_tile(const __bf16* gsrc,
                                                unsigned lds_byte_off,
                                                unsigned ldbt_elems) {
  unsigned long long ga = (unsigned long long)(uintptr_t)gsrc;
  u32x4 g0;
  g0[0] = 1u;                                   // count=1, user mode
  g0[1] = lds_byte_off;                         // lds_addr (bytes)
  g0[2] = (unsigned)ga;                         // global_addr[31:0]
  g0[3] = (unsigned)(ga >> 32) | (2u << 30);    // global_addr[56:32] | type=2
  u32x8 g1;
  g1[0] = 1u << 16;                             // data_size=1 (2 bytes)
  g1[1] = (ldbt_elems & 0xFFFFu) << 16;         // tensor_dim0[15:0]
  g1[2] = (ldbt_elems >> 16) | ((unsigned)BN << 16); // tensor_dim0[31:16] | tensor_dim1[15:0]
  g1[3] = ((unsigned)BK << 16);                 // tensor_dim1[31:16]=0 | tile_dim0
  g1[4] = (unsigned)BN;                         // tile_dim1 | tile_dim2=0
  g1[5] = ldbt_elems;                           // tensor_dim0_stride[31:0]
  g1[6] = 0u;                                   // stride0[47:32] | stride1[15:0]
  g1[7] = 0u;                                   // stride1[47:16]
  asm volatile("tensor_load_to_lds %0, %1, null, null"
               :: "s"(g0), "s"(g1) : "memory");
}

// ---------------------------------------------------------------------------
// Kernel: add embeddings + row LayerNorm (no affine; folded into Wkv) -> bf16
// ---------------------------------------------------------------------------
__global__ void prep_xhat_kernel(const float* __restrict__ x,
                                 const float* __restrict__ frame_emb,
                                 const float* __restrict__ media_emb,
                                 __bf16* __restrict__ xhat) {
  const int row = blockIdx.x;          // (b, t, f, v) flattened
  const int tid = threadIdx.x;         // 256
  const int f = (row >> 8) & 15;
  const int t = (row >> 12) & 7;
  const float* xr = x + (size_t)row * DIM;
  const float* fe = frame_emb + (size_t)f * DIM;
  const float* me = media_emb + (size_t)t * DIM;

  float v[4], s = 0.f, ss = 0.f;
#pragma unroll
  for (int i = 0; i < 4; ++i) {
    int c = tid + 256 * i;
    v[i] = xr[c] + fe[c] + me[c];
    s += v[i]; ss += v[i] * v[i];
  }
  __shared__ float sh_s[256], sh_ss[256];
  sh_s[tid] = s; sh_ss[tid] = ss; __syncthreads();
  for (int off = 128; off > 0; off >>= 1) {
    if (tid < off) { sh_s[tid] += sh_s[tid + off]; sh_ss[tid] += sh_ss[tid + off]; }
    __syncthreads();
  }
  const float mean = sh_s[0] * (1.f / DIM);
  const float var  = sh_ss[0] * (1.f / DIM) - mean * mean;
  const float rstd = rsqrtf(var + 1e-5f);
#pragma unroll
  for (int i = 0; i < 4; ++i) {
    int c = tid + 256 * i;
    xhat[(size_t)row * DIM + c] = (__bf16)((v[i] - mean) * rstd);
  }
}

// ---------------------------------------------------------------------------
// Kernel: generic row LayerNorm (with affine), f32 in, optional f32/bf16 out
// ---------------------------------------------------------------------------
__global__ void ln_rows_kernel(const float* __restrict__ in,
                               const float* __restrict__ w,
                               const float* __restrict__ b,
                               float* __restrict__ outF,
                               __bf16* __restrict__ outB) {
  const int row = blockIdx.x;
  const int tid = threadIdx.x;  // 256
  const float* xr = in + (size_t)row * DIM;
  float v[4], s = 0.f, ss = 0.f;
#pragma unroll
  for (int i = 0; i < 4; ++i) {
    v[i] = xr[tid + 256 * i];
    s += v[i]; ss += v[i] * v[i];
  }
  __shared__ float sh_s[256], sh_ss[256];
  sh_s[tid] = s; sh_ss[tid] = ss; __syncthreads();
  for (int off = 128; off > 0; off >>= 1) {
    if (tid < off) { sh_s[tid] += sh_s[tid + off]; sh_ss[tid] += sh_ss[tid + off]; }
    __syncthreads();
  }
  const float mean = sh_s[0] * (1.f / DIM);
  const float var  = sh_ss[0] * (1.f / DIM) - mean * mean;
  const float rstd = rsqrtf(var + 1e-5f);
#pragma unroll
  for (int i = 0; i < 4; ++i) {
    int c = tid + 256 * i;
    float y = (v[i] - mean) * rstd * w[c] + b[c];
    if (outF) outF[(size_t)row * DIM + c] = y;
    if (outB) outB[(size_t)row * DIM + c] = (__bf16)y;
  }
}

// ---------------------------------------------------------------------------
// Kernel: init latents, broadcast to (b*T, NLAT, DIM)
// ---------------------------------------------------------------------------
__global__ void init_lat_kernel(const float* __restrict__ latents,
                                float* __restrict__ lat) {
  const int row = blockIdx.x;        // 0..1023
  const int r = row & (NLAT - 1);
  const int tid = threadIdx.x;
#pragma unroll
  for (int i = 0; i < 4; ++i) {
    int c = tid + 256 * i;
    lat[(size_t)row * DIM + c] = latents[(size_t)r * DIM + c];
  }
}

// ---------------------------------------------------------------------------
// Kernel: convert f32 W[K][N] -> bf16 Wt[N][K] (transpose for WMMA B-frags)
// ---------------------------------------------------------------------------
__global__ void convt_kernel(const float* __restrict__ W,
                             __bf16* __restrict__ Wt, int K, int N) {
  int idx = blockIdx.x * 256 + threadIdx.x;
  if (idx >= K * N) return;
  int k = idx / N, n = idx - k * N;
  Wt[(size_t)n * K + k] = (__bf16)W[idx];
}

// ---------------------------------------------------------------------------
// Kernel: fold LN affine into Wkv; output transposed bf16 + bias row vector
//   Wt[n][k] = lnm_w[k] * Wkv[k][n];  bias[n] = sum_k lnm_b[k] * Wkv[k][n]
// ---------------------------------------------------------------------------
__global__ void fold_wkv_kernel(const float* __restrict__ Wkv,
                                const float* __restrict__ lw,
                                const float* __restrict__ lb,
                                __bf16* __restrict__ Wt,
                                float* __restrict__ biasv) {
  const int n = blockIdx.x;   // 0..1023
  const int tid = threadIdx.x;
  float part = 0.f;
  for (int k = tid; k < DIM; k += 256) {
    float wv = Wkv[(size_t)k * DIM + n];
    Wt[(size_t)n * DIM + k] = (__bf16)(lw[k] * wv);
    part += lb[k] * wv;
  }
  __shared__ float sh[256];
  sh[tid] = part; __syncthreads();
  for (int off = 128; off > 0; off >>= 1) {
    if (tid < off) sh[tid] += sh[tid + off];
    __syncthreads();
  }
  if (tid == 0) biasv[n] = sh[0];
}

// ---------------------------------------------------------------------------
// WMMA GEMM: C[M][N] = A[M][K] @ Bt[N][K]^T, bf16 in, f32 accumulate.
// Block = 256 threads (8 waves). Tile BMxBN = 128x64, K-chunks of 64,
// double-buffered LDS: A via async-to-LDS (ASYNCcnt), B via TDM (TENSORcnt).
// Wave w owns rows [w*16, w*16+16) x 64 cols: 4 accumulators; A frag reused
// across 4 WMMAs per K-step. Fragment layouts per ISA 7.12.2.
// Epilogue: v = acc*scale (+bias[col]) (gelu) (+resid) -> f32 and/or bf16.
// flags: bit0 = exact GELU
// ---------------------------------------------------------------------------
__global__ void gemm_wmma_kernel(const __bf16* __restrict__ A,
                                 const __bf16* __restrict__ Bt,
                                 float* __restrict__ outF,
                                 __bf16* __restrict__ outB,
                                 const float* __restrict__ bias,
                                 const float* __restrict__ resid,
                                 int K, int lda, int ldbt, int ldo, int ldr,
                                 float scale, int flags) {
  __shared__ __bf16 smem[2][A_ELEMS + B_ELEMS];

  const int tid  = threadIdx.x;
  const int lane = tid & 31;
  const int wave = tid >> 5;
  const int half = lane >> 4;
  const int l16  = lane & 15;
  const int row0 = blockIdx.y * BM;
  const int col0 = blockIdx.x * BN;
  const int NC = K / BK;
  // Wave-uniform SGPR predicate -> scalar branch. TDM issues regardless of
  // EXEC, so an exec-masked guard would make all 8 waves issue duplicate DMAs.
  const bool w0 = (__builtin_amdgcn_readfirstlane(tid) >> 5) == 0;

  auto issueA = [&](int c, int b) {
    const int k0 = c * BK;
#pragma unroll
    for (int i = 0; i < 4; ++i) {
      int idx = i * 256 + tid;          // 0..1023
      int r = idx >> 3, seg = idx & 7;  // row in [0,128), 16B segment in [0,8)
      async_copy_b128(A + (size_t)(row0 + r) * lda + k0 + seg * 8,
                      &smem[b][r * BK + seg * 8]);
    }
  };
  auto issueB = [&](int c, int b) {     // one TDM DMA for the whole B panel
    tdm_load_b_tile(Bt + (size_t)col0 * ldbt + c * BK,
                    (unsigned)(uintptr_t)&smem[b][A_ELEMS], (unsigned)ldbt);
  };

  issueA(0, 0);
  if (w0) issueB(0, 0);

  v8f acc0 = {}, acc1 = {}, acc2 = {}, acc3 = {};
  int buf = 0;

  for (int c = 0; c < NC; ++c) {
    if (c + 1 < NC) {
      issueA(c + 1, buf ^ 1);
      if (w0) issueB(c + 1, buf ^ 1);
      wait_async_le4();                 // current A chunk retired (in-order)
      if (w0) __builtin_amdgcn_s_wait_tensorcnt(1);
    } else {
      wait_async_0();
      if (w0) __builtin_amdgcn_s_wait_tensorcnt(0);
    }
    __syncthreads();                    // all waves' current chunk in LDS

    const __bf16* As = &smem[buf][0];
    const __bf16* Bs = &smem[buf][A_ELEMS];
#pragma unroll
    for (int ks = 0; ks < BK; ks += 32) {
      // A fragment: row = wave*16 + l16; two contiguous 8-elem runs per half
      const __bf16* ar = As + (wave * 16 + l16) * BK + ks + half * 8;
      v8bf alo = *(const v8bf*)(ar);
      v8bf ahi = *(const v8bf*)(ar + 16);
      v16bf af;
#pragma unroll
      for (int i = 0; i < 8; ++i) { af[i] = alo[i]; af[8 + i] = ahi[i]; }
      // preload all 4 B fragments, then issue 4 WMMAs back-to-back
      v16bf bfr[4];
#pragma unroll
      for (int nt = 0; nt < 4; ++nt) {
        const __bf16* br = Bs + (nt * 16 + l16) * BK + ks + half * 8;
        v8bf blo = *(const v8bf*)(br);
        v8bf bhi = *(const v8bf*)(br + 16);
#pragma unroll
        for (int i = 0; i < 8; ++i) { bfr[nt][i] = blo[i]; bfr[nt][8 + i] = bhi[i]; }
      }
      acc0 = __builtin_amdgcn_wmma_f32_16x16x32_bf16(false, af, false, bfr[0], (short)0, acc0, false, false);
      acc1 = __builtin_amdgcn_wmma_f32_16x16x32_bf16(false, af, false, bfr[1], (short)0, acc1, false, false);
      acc2 = __builtin_amdgcn_wmma_f32_16x16x32_bf16(false, af, false, bfr[2], (short)0, acc2, false, false);
      acc3 = __builtin_amdgcn_wmma_f32_16x16x32_bf16(false, af, false, bfr[3], (short)0, acc3, false, false);
    }
    __syncthreads();                    // everyone done reading buf
    buf ^= 1;
  }

  // epilogue: C layout per tile: row = r + half*8, col = l16
#pragma unroll
  for (int nt = 0; nt < 4; ++nt) {
    const v8f acc = (nt == 0) ? acc0 : (nt == 1) ? acc1 : (nt == 2) ? acc2 : acc3;
    const int col = col0 + nt * 16 + l16;
#pragma unroll
    for (int r = 0; r < 8; ++r) {
      const int row = row0 + wave * 16 + (half << 3) + r;
      float v = acc[r] * scale;
      if (bias)      v += bias[col];
      if (flags & 1) v = 0.5f * v * (1.0f + erff(v * 0.70710678118654752f));
      if (resid)     v += resid[(size_t)row * ldr + col];
      if (outF) outF[(size_t)row * ldo + col] = v;
      if (outB) outB[(size_t)row * ldo + col] = (__bf16)v;
    }
  }
}

// ---------------------------------------------------------------------------
// Attention: one block per (bt, head); 64 threads, one per latent row.
// Online softmax over 4160 keys (4096 media + 64 latent), K/V bf16.
// q already scaled by DH^-0.5. o written bf16 at column h*DH+dh.
// ---------------------------------------------------------------------------
__global__ void attn_kernel(const float* __restrict__ q,
                            const __bf16* __restrict__ kvx,
                            const __bf16* __restrict__ kvlat,
                            __bf16* __restrict__ o) {
  const int h  = blockIdx.x & (NHEAD - 1);
  const int bt = blockIdx.x >> 3;
  const int r  = threadIdx.x;        // 0..63 latent row

  const float* qr = q + ((size_t)(bt * NLAT + r)) * INNER + h * DH;
  float qv[DH], accv[DH];
#pragma unroll
  for (int d = 0; d < DH; ++d) { qv[d] = qr[d]; accv[d] = 0.f; }

  float m = -3.0e38f, l = 0.f;
  for (int j = 0; j < NMEDIA + NLAT; ++j) {
    const __bf16* kr = (j < NMEDIA)
        ? kvx   + ((size_t)bt * NMEDIA + j) * DIM + h * DH
        : kvlat + ((size_t)bt * NLAT + (j - NMEDIA)) * DIM + h * DH;
    const __bf16* vr = kr + INNER;   // V at cols 512 + h*64, same row
    float s = 0.f;
#pragma unroll
    for (int d = 0; d < DH; ++d) s += qv[d] * (float)kr[d];
    const float mn   = fmaxf(m, s);
    const float corr = __expf(m - mn);
    const float p    = __expf(s - mn);
    l = l * corr + p;
#pragma unroll
    for (int d = 0; d < DH; ++d) accv[d] = accv[d] * corr + p * (float)vr[d];
    m = mn;
  }
  __bf16* orow = o + ((size_t)(bt * NLAT + r)) * INNER + h * DH;
  const float inv = 1.f / l;
#pragma unroll
  for (int d = 0; d < DH; ++d) orow[d] = (__bf16)(accv[d] * inv);
}

// ---------------------------------------------------------------------------
// Host orchestration
// ---------------------------------------------------------------------------
static inline void launch_gemm(const __bf16* A, const __bf16* Bt,
                               float* outF, __bf16* outB,
                               const float* bias, const float* resid,
                               int M, int N, int K, int lda, int ldbt,
                               int ldo, int ldr, float scale, int flags,
                               hipStream_t stream) {
  dim3 grid(N / BN, M / BM);
  gemm_wmma_kernel<<<grid, 256, 0, stream>>>(A, Bt, outF, outB, bias, resid,
                                             K, lda, ldbt, ldo, ldr, scale, flags);
}

extern "C" void kernel_launch(void* const* d_in, const int* in_sizes, int n_in,
                              void* d_out, int out_size, void* d_ws, size_t ws_size,
                              hipStream_t stream) {
  (void)in_sizes; (void)n_in; (void)out_size; (void)ws_size;
  const float* x         = (const float*)d_in[0];
  const float* latents   = (const float*)d_in[1];
  const float* frame_emb = (const float*)d_in[2];
  const float* media_emb = (const float*)d_in[3];
  const float* lnm_w     = (const float*)d_in[4];
  const float* lnm_b     = (const float*)d_in[5];
  const float* lnl_w     = (const float*)d_in[6];
  const float* lnl_b     = (const float*)d_in[7];
  const float* Wq        = (const float*)d_in[8];
  const float* Wkv       = (const float*)d_in[9];
  const float* Wo        = (const float*)d_in[10];
  const float* ffln_w    = (const float*)d_in[11];
  const float* ffln_b    = (const float*)d_in[12];
  const float* W1        = (const float*)d_in[13];
  const float* b1        = (const float*)d_in[14];
  const float* W2        = (const float*)d_in[15];
  const float* b2        = (const float*)d_in[16];
  const float* fln_w     = (const float*)d_in[17];
  const float* fln_b     = (const float*)d_in[18];

  // ---- workspace carve ----
  uint8_t* ws = (uint8_t*)d_ws;
  size_t off = 0;
  auto carve = [&](size_t bytes) -> void* {
    void* p = ws + off;
    off += (bytes + 255) & ~(size_t)255;
    return p;
  };
  __bf16* xhat      = (__bf16*)carve((size_t)XROWS * DIM * 2);   // 134 MB
  __bf16* kvx       = (__bf16*)carve((size_t)XROWS * DIM * 2);   // 134 MB
  float*  lat       = (float*) carve((size_t)LROWS * DIM * 4);   // 4 MB
  __bf16* latn_bf   = (__bf16*)carve((size_t)LROWS * DIM * 2);
  float*  q_f32     = (float*) carve((size_t)LROWS * INNER * 4);
  __bf16* kvlat_bf  = (__bf16*)carve((size_t)LROWS * DIM * 2);
  __bf16* o_bf      = (__bf16*)carve((size_t)LROWS * INNER * 2);
  __bf16* h_bf      = (__bf16*)carve((size_t)LROWS * DIM * 2);
  __bf16* g_bf      = (__bf16*)carve((size_t)LROWS * FFDIM * 2);
  __bf16* Wq_t      = (__bf16*)carve((size_t)INNER * DIM * 2);
  __bf16* Wkv_t     = (__bf16*)carve((size_t)DIM * DIM * 2);
  __bf16* Wkv_eff_t = (__bf16*)carve((size_t)DIM * DIM * 2);
  float*  bias_kv   = (float*) carve((size_t)DIM * 4);
  __bf16* Wo_t      = (__bf16*)carve((size_t)DIM * INNER * 2);
  __bf16* W1_t      = (__bf16*)carve((size_t)FFDIM * DIM * 2);
  __bf16* W2_t      = (__bf16*)carve((size_t)DIM * FFDIM * 2);

  // ---- one-time prep ----
  prep_xhat_kernel<<<XROWS, 256, 0, stream>>>(x, frame_emb, media_emb, xhat);
  init_lat_kernel<<<LROWS, 256, 0, stream>>>(latents, lat);

  const float qscale = 0.125f;   // DH^-0.5

  for (int l = 0; l < DEPTH; ++l) {
    const float* Wq_l  = Wq  + (size_t)l * DIM * INNER;
    const float* Wkv_l = Wkv + (size_t)l * DIM * DIM;
    const float* Wo_l  = Wo  + (size_t)l * INNER * DIM;
    const float* W1_l  = W1  + (size_t)l * DIM * FFDIM;
    const float* W2_l  = W2  + (size_t)l * FFDIM * DIM;

    // weight conversion / fold (bf16, N-major)
    convt_kernel<<<(DIM * INNER) / 256, 256, 0, stream>>>(Wq_l, Wq_t, DIM, INNER);
    convt_kernel<<<(DIM * DIM) / 256, 256, 0, stream>>>(Wkv_l, Wkv_t, DIM, DIM);
    fold_wkv_kernel<<<DIM, 256, 0, stream>>>(Wkv_l, lnm_w + l * DIM, lnm_b + l * DIM,
                                             Wkv_eff_t, bias_kv);
    convt_kernel<<<(INNER * DIM) / 256, 256, 0, stream>>>(Wo_l, Wo_t, INNER, DIM);
    convt_kernel<<<(DIM * FFDIM) / 256, 256, 0, stream>>>(W1_l, W1_t, DIM, FFDIM);
    convt_kernel<<<(FFDIM * DIM) / 256, 256, 0, stream>>>(W2_l, W2_t, FFDIM, DIM);

    // ln_lat -> bf16
    ln_rows_kernel<<<LROWS, 256, 0, stream>>>(lat, lnl_w + l * DIM, lnl_b + l * DIM,
                                              nullptr, latn_bf);

    // q = (ln_lat @ Wq) * scale    [1024 x 512] f32
    launch_gemm(latn_bf, Wq_t, q_f32, nullptr, nullptr, nullptr,
                LROWS, INNER, DIM, DIM, DIM, INNER, 0, qscale, 0, stream);
    // kv_lat = ln_lat @ Wkv        [1024 x 1024] bf16
    launch_gemm(latn_bf, Wkv_t, nullptr, kvlat_bf, nullptr, nullptr,
                LROWS, DIM, DIM, DIM, DIM, DIM, 0, 1.0f, 0, stream);
    // kv_x = xhat @ Wkv_eff + bias [65536 x 1024] bf16  (dominant GEMM)
    launch_gemm(xhat, Wkv_eff_t, nullptr, kvx, bias_kv, nullptr,
                XROWS, DIM, DIM, DIM, DIM, DIM, 0, 1.0f, 0, stream);

    // attention (online softmax over 4160 keys)
    attn_kernel<<<NBT * NHEAD, NLAT, 0, stream>>>(q_f32, kvx, kvlat_bf, o_bf);

    // lat = o @ Wo + lat
    launch_gemm(o_bf, Wo_t, lat, nullptr, nullptr, lat,
                LROWS, DIM, INNER, INNER, INNER, DIM, DIM, 1.0f, 0, stream);

    // FF: h = ln(lat) ; g = gelu(h@W1+b1) ; lat = g@W2+b2 + lat
    ln_rows_kernel<<<LROWS, 256, 0, stream>>>(lat, ffln_w + l * DIM, ffln_b + l * DIM,
                                              nullptr, h_bf);
    launch_gemm(h_bf, W1_t, nullptr, g_bf, b1 + (size_t)l * FFDIM, nullptr,
                LROWS, FFDIM, DIM, DIM, DIM, FFDIM, 0, 1.0f, /*gelu*/1, stream);
    launch_gemm(g_bf, W2_t, lat, nullptr, b2 + (size_t)l * DIM, lat,
                LROWS, DIM, FFDIM, FFDIM, FFDIM, DIM, DIM, 1.0f, 0, stream);
  }

  // final LN -> f32 output (b, T, NLAT, DIM)
  ln_rows_kernel<<<LROWS, 256, 0, stream>>>(lat, fln_w, fln_b, (float*)d_out, nullptr);
}
